// HippocampalMemory_27212912787968
// MI455X (gfx1250) — matrix operand order repaired and checked by reading, not compiled, not run on hardware
//
#include <hip/hip_runtime.h>
#include <hip/hip_bf16.h>

#define BQ      256     // batch of queries
#define DMODEL  1024    // d_model
#define DGDIM   4096    // dg expansion dim
#define MSZ     65536   // memory size
#define TOPKN   32
#define SPARSEK 81
#define KC      128     // K-chunk staged in LDS for sims GEMM
#define LDB     136     // padded LDS row stride (bf16): 128 + 8 -> conflict-free b128 reads

typedef __attribute__((ext_vector_type(16))) __bf16 v16bf;
typedef __attribute__((ext_vector_type(8)))  __bf16 v8bf;
typedef __attribute__((ext_vector_type(8)))  float  v8f;

// ---- WMMA A fragment (bf16 16x32, M x K), row-major source, ld = ldk ----
// lane<16: row = Rm+lane,   K = [kc..kc+15]
// lane>=16: row = Rm+lane-16, K = [kc+16..kc+31]
static __device__ __forceinline__ v16bf load_a_frag(const __bf16* __restrict__ A,
                                                    int Rm, int kc, int lane, int ldk) {
  int r  = Rm + (lane & 15);
  int ko = kc + ((lane >> 4) << 4);
  const __bf16* p = A + (size_t)r * ldk + ko;
  v8bf lo = *(const v8bf*)p;          // 16B: K ko..ko+7
  v8bf hi = *(const v8bf*)(p + 8);    // 16B: K ko+8..ko+15
  v16bf a;
#pragma unroll
  for (int i = 0; i < 8; ++i) { a[i] = lo[i]; a[8 + i] = hi[i]; }
  return a;
}

// ---- WMMA B fragment (bf16 32x16, K x N) from fp32 row-major global source ----
static __device__ __forceinline__ v16bf load_b_frag_f32(const float* __restrict__ Src,
                                                        int col, int kc, int lane, int ldk) {
  int ko = kc + ((lane >> 4) << 4);
  const float4* p = (const float4*)(Src + (size_t)col * ldk + ko);
  float4 f[4];
#pragma unroll
  for (int i = 0; i < 4; ++i) f[i] = p[i];
  v16bf b;
#pragma unroll
  for (int i = 0; i < 4; ++i) {
    b[4 * i + 0] = (__bf16)f[i].x; b[4 * i + 1] = (__bf16)f[i].y;
    b[4 * i + 2] = (__bf16)f[i].z; b[4 * i + 3] = (__bf16)f[i].w;
  }
  return b;
}

// ---- WMMA B fragment from LDS-staged bf16 tile (16 rows x KC, stride LDB) ----
static __device__ __forceinline__ v16bf load_b_frag_lds(const __bf16* tile, int kk2, int lane) {
  const __bf16* bp = tile + (lane & 15) * LDB + kk2 + ((lane >> 4) << 4);
  v8bf lo = *(const v8bf*)bp;
  v8bf hi = *(const v8bf*)(bp + 8);
  v16bf b;
#pragma unroll
  for (int i = 0; i < 8; ++i) { b[i] = lo[i]; b[8 + i] = hi[i]; }
  return b;
}

// ---------------- utility kernels ----------------
__global__ void k_cvt_bf16(const float* __restrict__ in, __bf16* __restrict__ out, int n) {
  int i = blockIdx.x * blockDim.x + threadIdx.x;
  if (i < n) out[i] = (__bf16)in[i];
}
__global__ void k_copy_f32(const float* __restrict__ in, float* __restrict__ out, int n) {
  int i = blockIdx.x * blockDim.x + threadIdx.x;
  if (i < n) out[i] = in[i];
}

// ---------------- GEMM1: expanded = relu(q @ W^T + b) ----------------
// grid = DGDIM/16 blocks, 256 threads (8 waves, 2 M-tiles each -> 256 rows)
__global__ __launch_bounds__(256) void k_dg_gemm(const __bf16* __restrict__ qbf,
                                                 const float* __restrict__ W,
                                                 const float* __restrict__ bias,
                                                 float* __restrict__ expanded) {
  const int lane  = threadIdx.x & 31;
  const int wave  = threadIdx.x >> 5;
  const int nbase = blockIdx.x << 4;
  const int col   = nbase + (lane & 15);
  const int Rm0   = wave << 5, Rm1 = Rm0 + 16;
  v8f acc0 = {}, acc1 = {};
  for (int kc = 0; kc < DMODEL; kc += 32) {
    v16bf bf = load_b_frag_f32(W, col, kc, lane, DMODEL);
    v16bf a0 = load_a_frag(qbf, Rm0, kc, lane, DMODEL);
    v16bf a1 = load_a_frag(qbf, Rm1, kc, lane, DMODEL);
    acc0 = __builtin_amdgcn_wmma_f32_16x16x32_bf16(false, a0, false, bf, (short)0, acc0, false, false);
    acc1 = __builtin_amdgcn_wmma_f32_16x16x32_bf16(false, a1, false, bf, (short)0, acc1, false, false);
  }
  const float bn   = bias[col];
  const int   rofs = (lane >> 4) << 3;
#pragma unroll
  for (int i = 0; i < 8; ++i) {
    int r0 = Rm0 + rofs + i;
    float v0 = acc0[i] + bn; v0 = v0 > 0.f ? v0 : 0.f;
    expanded[(size_t)r0 * DGDIM + col] = v0;
    int r1 = Rm1 + rofs + i;
    float v1 = acc1[i] + bn; v1 = v1 > 0.f ? v1 : 0.f;
    expanded[(size_t)r1 * DGDIM + col] = v1;
  }
}

// ------- exact per-row top-81 threshold (radix select on float bits) -------
// values >= 0 (post-ReLU) so raw IEEE bits are monotone.
__global__ __launch_bounds__(256) void k_sparse_select(const float* __restrict__ expanded,
                                                       __bf16* __restrict__ sqbf,
                                                       float* __restrict__ rq) {
  const int b = blockIdx.x, tid = threadIdx.x;
  __shared__ unsigned hist[256];
  __shared__ unsigned s_sel;
  __shared__ int s_kk;
  __shared__ float red[256];

  float v[16]; unsigned u[16];
  const float* row = expanded + (size_t)b * DGDIM;
#pragma unroll
  for (int j = 0; j < 16; ++j) {
    float f = row[j * 256 + tid];
    v[j] = f; u[j] = __float_as_uint(f);
  }
  unsigned prefix = 0, prefmask = 0;
  int kk = SPARSEK;
  for (int pass = 3; pass >= 0; --pass) {
    hist[tid] = 0;
    __syncthreads();
    const int sh = pass << 3;
#pragma unroll
    for (int j = 0; j < 16; ++j)
      if ((u[j] & prefmask) == prefix) atomicAdd(&hist[(u[j] >> sh) & 0xFF], 1u);
    __syncthreads();
    if (tid == 0) {
      int cum = 0, d = 255;
      for (; d > 0; --d) { int c = (int)hist[d]; if (cum + c >= kk) break; cum += c; }
      s_sel = (unsigned)d; s_kk = kk - cum;
    }
    __syncthreads();
    prefix |= s_sel << sh;
    prefmask |= 0xFFu << sh;
    kk = s_kk;
    __syncthreads();
  }
  // prefix == bit pattern of the 81st largest value; keep v >= thr (tie-inclusive)
  float ssq = 0.f;
  __bf16* orow = sqbf + (size_t)b * DGDIM;
#pragma unroll
  for (int j = 0; j < 16; ++j) {
    float val = (u[j] >= prefix) ? v[j] : 0.f;
    orow[j * 256 + tid] = (__bf16)val;
    ssq += val * val;
  }
  red[tid] = ssq;
  __syncthreads();
  for (int s = 128; s > 0; s >>= 1) {
    if (tid < s) red[tid] += red[tid + s];
    __syncthreads();
  }
  if (tid == 0) rq[b] = 1.f / fmaxf(sqrtf(red[0]), 1e-8f);
}

// ---- GEMM2: sims = (sq @ keys^T) * importance / (|sq| * |key|) ----
// grid = MSZ/16 blocks. Key chunk (16 x KC fp32) staged cooperatively:
// loaded once, sum-of-squares fused, converted to bf16 once into a padded LDS
// tile shared by all 8 waves. Next chunk's global loads issued before compute.
__global__ __launch_bounds__(256) void k_sims_gemm(const __bf16* __restrict__ sqbf,
                                                   const float* __restrict__ keys,
                                                   const float* __restrict__ importance,
                                                   const float* __restrict__ rq,
                                                   float* __restrict__ sims) {
  __shared__ __bf16 tile[16 * LDB];
  __shared__ float red[256];
  __shared__ float rk_sh[16];

  const int tid   = threadIdx.x;
  const int lane  = tid & 31;
  const int wave  = tid >> 5;
  const int nbase = blockIdx.x << 4;
  const int Rm0   = wave << 5, Rm1 = Rm0 + 16;

  // cooperative staging role: 16 threads per key row, 8 floats each
  const int kr  = tid >> 4;   // key row 0..15
  const int seg = tid & 15;   // segment 0..15 -> K offset seg*8
  const float* kp_base = keys + (size_t)(nbase + kr) * DGDIM + seg * 8;

  v8f acc0 = {}, acc1 = {};
  float ss = 0.f;

  float4 f0 = *(const float4*)(kp_base + 0);
  float4 f1 = *(const float4*)(kp_base + 4);

  for (int kc = 0; kc < DGDIM; kc += KC) {
    // fused sum-of-squares (fp32, full precision) + bf16 convert + LDS store
    ss += f0.x * f0.x + f0.y * f0.y + f0.z * f0.z + f0.w * f0.w
        + f1.x * f1.x + f1.y * f1.y + f1.z * f1.z + f1.w * f1.w;
    v8bf pk;
    pk[0] = (__bf16)f0.x; pk[1] = (__bf16)f0.y; pk[2] = (__bf16)f0.z; pk[3] = (__bf16)f0.w;
    pk[4] = (__bf16)f1.x; pk[5] = (__bf16)f1.y; pk[6] = (__bf16)f1.z; pk[7] = (__bf16)f1.w;
    *(v8bf*)(tile + kr * LDB + seg * 8) = pk;
    __syncthreads();

    if (kc + KC < DGDIM) {        // uniform branch: prefetch next chunk during compute
      f0 = *(const float4*)(kp_base + kc + KC + 0);
      f1 = *(const float4*)(kp_base + kc + KC + 4);
    }

#pragma unroll
    for (int kk2 = 0; kk2 < KC; kk2 += 32) {
      v16bf bf = load_b_frag_lds(tile, kk2, lane);
      v16bf a0 = load_a_frag(sqbf, Rm0, kc + kk2, lane, DGDIM);
      v16bf a1 = load_a_frag(sqbf, Rm1, kc + kk2, lane, DGDIM);
      acc0 = __builtin_amdgcn_wmma_f32_16x16x32_bf16(false, a0, false, bf, (short)0, acc0, false, false);
      acc1 = __builtin_amdgcn_wmma_f32_16x16x32_bf16(false, a1, false, bf, (short)0, acc1, false, false);
    }
    __syncthreads();
  }

  // per-key norm: reduce the 16 staging threads of each key row
  red[tid] = ss;
  __syncthreads();
  if (tid < 16) {
    float s = 0.f;
#pragma unroll
    for (int i = 0; i < 16; ++i) s += red[tid * 16 + i];
    float kn = fmaxf(sqrtf(s), 1e-8f);
    rk_sh[tid] = importance[nbase + tid] / kn;
  }
  __syncthreads();

  const float rk  = rk_sh[lane & 15];
  const int   key = nbase + (lane & 15);
  const int  rofs = (lane >> 4) << 3;
#pragma unroll
  for (int i = 0; i < 8; ++i) {
    int r0 = Rm0 + rofs + i;
    sims[(size_t)r0 * MSZ + key] = acc0[i] * rq[r0] * rk;
    int r1 = Rm1 + rofs + i;
    sims[(size_t)r1 * MSZ + key] = acc1[i] * rq[r1] * rk;
  }
}

// ---- per-row top-32: signed-float radix select + ordered extraction ----
static __device__ __forceinline__ unsigned fmap(float f) {
  unsigned b = __float_as_uint(f);
  return (b & 0x80000000u) ? ~b : (b | 0x80000000u);
}

__global__ __launch_bounds__(256) void k_topk(const float* __restrict__ sims,
                                              float* __restrict__ top_sim,
                                              int* __restrict__ top_idx,
                                              float* __restrict__ usage) {
  const int b = blockIdx.x, tid = threadIdx.x;
  __shared__ unsigned hist[256];
  __shared__ unsigned s_sel;
  __shared__ int s_kk;
  __shared__ int cnt;
  __shared__ float cv[64];
  __shared__ int ci[64];
  const float* row = sims + (size_t)b * MSZ;

  unsigned prefix = 0, prefmask = 0;
  int kk = TOPKN;
  for (int pass = 3; pass >= 0; --pass) {
    hist[tid] = 0;
    __syncthreads();
    const int sh = pass << 3;
    for (int j = 0; j < MSZ / 256; ++j) {
      unsigned u = fmap(row[j * 256 + tid]);
      if ((u & prefmask) == prefix) atomicAdd(&hist[(u >> sh) & 0xFF], 1u);
    }
    __syncthreads();
    if (tid == 0) {
      int cum = 0, d = 255;
      for (; d > 0; --d) { int c = (int)hist[d]; if (cum + c >= kk) break; cum += c; }
      s_sel = (unsigned)d; s_kk = kk - cum;
    }
    __syncthreads();
    prefix |= s_sel << sh;
    prefmask |= 0xFFu << sh;
    kk = s_kk;
    __syncthreads();
  }
  if (tid == 0) cnt = 0;
  __syncthreads();
  for (int j = 0; j < MSZ / 256; ++j) {
    int idx = j * 256 + tid;
    float s = row[idx];
    if (fmap(s) >= prefix) {
      int p = atomicAdd(&cnt, 1);
      if (p < 64) { cv[p] = s; ci[p] = idx; }
    }
  }
  __syncthreads();
  if (tid == 0) {
    int n = cnt < 64 ? cnt : 64;
    for (int j = 0; j < TOPKN; ++j) {     // value desc, index asc (lax.top_k order)
      float bv = -3.4e38f; int bi = 0x7FFFFFFF; int bp = -1;
      for (int i = 0; i < n; ++i) {
        float vv = cv[i]; int ii = ci[i];
        if (ii >= 0 && (vv > bv || (vv == bv && ii < bi))) { bv = vv; bi = ii; bp = i; }
      }
      if (bp >= 0) {
        ci[bp] = -1;
        top_sim[b * TOPKN + j] = bv;
        top_idx[b * TOPKN + j] = bi;
        atomicAdd(&usage[bi], 1.0f);
      } else {
        top_sim[b * TOPKN + j] = 0.f;
        top_idx[b * TOPKN + j] = 0;
      }
    }
  }
}

// ---- gather retrieved rows: one block per (b, j), float4 vectorized ----
__global__ __launch_bounds__(256) void k_gather(const float* __restrict__ values,
                                                const int* __restrict__ top_idx,
                                                float* __restrict__ retrieved) {
  const int blk = blockIdx.x;              // b*32 + j
  const int idx = top_idx[blk];
  const float4* src = (const float4*)(values + (size_t)idx * DMODEL);
  float4* dst = (float4*)(retrieved + (size_t)blk * DMODEL);
  dst[threadIdx.x] = src[threadIdx.x];     // 256 threads * 4 floats = 1024
}

extern "C" void kernel_launch(void* const* d_in, const int* in_sizes, int n_in,
                              void* d_out, int out_size, void* d_ws, size_t ws_size,
                              hipStream_t stream) {
  (void)in_sizes; (void)n_in; (void)out_size; (void)ws_size;
  const float* query    = (const float*)d_in[0];   // (256,1024)
  const float* dg_W     = (const float*)d_in[1];   // (4096,1024)
  const float* dg_b     = (const float*)d_in[2];   // (4096,)
  const float* keys     = (const float*)d_in[3];   // (65536,4096)
  const float* values   = (const float*)d_in[4];   // (65536,1024)
  const float* imp      = (const float*)d_in[5];   // (65536,)
  const float* usage_in = (const float*)d_in[6];   // (65536,)

  char* ws = (char*)d_ws;
  __bf16* q_bf   = (__bf16*)(ws + 0);                      // 512 KB
  float*  expand = (float*)(ws + (1u << 20));              // 4 MB
  __bf16* sq_bf  = (__bf16*)(ws + (5u << 20));             // 2 MB
  float*  rq     = (float*)(ws + (7u << 20));              // 1 KB
  int*    tidx   = (int*)(ws + (7u << 20) + 4096);         // 32 KB
  float*  sims   = (float*)(ws + (8u << 20));              // 64 MB (L2-resident)

  float* out_ret = (float*)d_out;                          // (256,32,1024)
  float* out_sim = out_ret + (size_t)BQ * TOPKN * DMODEL;  // (256,32)
  float* out_use = out_sim + (size_t)BQ * TOPKN;           // (65536,)

  k_cvt_bf16<<<(BQ * DMODEL + 255) / 256, 256, 0, stream>>>(query, q_bf, BQ * DMODEL);
  k_copy_f32<<<(MSZ + 255) / 256, 256, 0, stream>>>(usage_in, out_use, MSZ);
  k_dg_gemm<<<DGDIM / 16, 256, 0, stream>>>(q_bf, dg_W, dg_b, expand);
  k_sparse_select<<<BQ, 256, 0, stream>>>(expand, sq_bf, rq);
  k_sims_gemm<<<MSZ / 16, 256, 0, stream>>>(sq_bf, keys, imp, rq, sims);
  k_topk<<<BQ, 256, 0, stream>>>(sims, out_sim, tidx, out_use);
  k_gather<<<BQ * TOPKN, 256, 0, stream>>>(values, tidx, out_ret);
}
